// GhebConvV1_16020228014638
// MI455X (gfx1250) — matrix operand order repaired and checked by reading, not compiled, not run on hardware
//
#include <hip/hip_runtime.h>
#include <hip/hip_bf16.h>

// ---------------------------------------------------------------------------
// ChebConv GNN for MI455X (gfx1250): sparse propagate (L2-resident atomics) +
// fused per-layer GEMM on v_wmma_f32_16x16x32_bf16 with fragment-major packing
// and async global->LDS double-buffered B-tile staging (ASYNCcnt path).
// ---------------------------------------------------------------------------

#define NN      10000
#define NPAD    10048          // multiple of 64 for the GEMM tiling
#define EE      320000
#define NG      64
#define KCHEB   10
#define INC     128
#define HIDC    256
#define OUTC    128
#define KT_MAX  ((KCHEB * HIDC) / 32)   // 80
#define MT_CNT  (NPAD / 16)             // 628

typedef __attribute__((ext_vector_type(16))) __bf16 v16bf;
typedef __attribute__((ext_vector_type(8)))  float  v8f;

__device__ __forceinline__ unsigned short f2bf(float f) {
  unsigned u = __float_as_uint(f);
  u += 0x7FFFu + ((u >> 16) & 1u);      // round-to-nearest-even
  return (unsigned short)(u >> 16);
}

// ------------------------------- utility ----------------------------------
__global__ void k_zero_f32(float* __restrict__ p, int n) {
  int i = blockIdx.x * blockDim.x + threadIdx.x;
  if (i < n) p[i] = 0.f;
}

// ------------------------- edge norm (deg, weights) ------------------------
__global__ void k_deg(const int* __restrict__ src, float* __restrict__ deg, int E) {
  int e = blockIdx.x * blockDim.x + threadIdx.x;
  if (e < E) atomicAdd(&deg[src[e]], 1.0f);
}

__global__ void k_edgew(const int* __restrict__ src, const int* __restrict__ dst,
                        const float* __restrict__ deg, float* __restrict__ w, int E) {
  int e = blockIdx.x * blockDim.x + threadIdx.x;
  if (e >= E) return;
  float ds = deg[src[e]], dd = deg[dst[e]];
  float a = ds > 0.f ? rsqrtf(fmaxf(ds, 1.f)) : 0.f;
  float b = dd > 0.f ? rsqrtf(fmaxf(dd, 1.f)) : 0.f;
  w[e] = -a * b;
}

// --------------------------- sparse propagate ------------------------------
// y[dst] += w_e * x[src]; one thread per (edge, 4-channel group).
// x (10MB) lives in the 192MB L2, so gathers + f32 atomics stay on-chip.
__global__ void k_propagate(const float* __restrict__ x, const int* __restrict__ src,
                            const int* __restrict__ dst, const float* __restrict__ w,
                            float* __restrict__ y, int E, int c4Shift) {
  int i = blockIdx.x * blockDim.x + threadIdx.x;
  int e = i >> c4Shift;
  if (e >= E) return;
  int cg = i & ((1 << c4Shift) - 1);
  int C = 4 << c4Shift;
  float we = w[e];
  const float4 v = ((const float4*)(x + (size_t)src[e] * C))[cg];
  float* yd = y + (size_t)dst[e] * C + cg * 4;
  atomicAdd(yd + 0, we * v.x);
  atomicAdd(yd + 1, we * v.y);
  atomicAdd(yd + 2, we * v.z);
  atomicAdd(yd + 3, we * v.w);
}

// Chebyshev recurrence: out = 2*y - a (in-place-safe elementwise)
__global__ void k_cheb(const float* __restrict__ y, const float* __restrict__ a,
                       float* __restrict__ out, int n) {
  int i = blockIdx.x * blockDim.x + threadIdx.x;
  if (i < n) out[i] = 2.f * y[i] - a[i];
}

// ------------------------ fragment-major packing ---------------------------
// A (16x32 bf16 tile) per ISA 7.12.2: lane = (m&15) | (grp<<4), grp=(kk>>3)&1,
// elem j = (kk&7) | ((kk>>4)<<3).   Each lane's fragment = contiguous 32B.
__global__ void k_pack_a(const float* __restrict__ src, unsigned short* __restrict__ Ap,
                         int nValid, int cShift, int KT, int kOff) {
  int i = blockIdx.x * blockDim.x + threadIdx.x;
  int C = 1 << cShift;
  int total = NPAD << cShift;
  if (i >= total) return;
  int m = i >> cShift;
  int c = i & (C - 1);
  float v = (m < nValid) ? src[(size_t)m * C + c] : 0.f;
  int kg = kOff + c;
  int kt = kg >> 5, kk = kg & 31;
  int grp = (kk >> 3) & 1;
  int j = (kk & 7) | ((kk >> 4) << 3);
  int lane = (m & 15) | (grp << 4);
  size_t d = (((size_t)(m >> 4) * KT + kt) * 32 + lane) * 16 + j;
  Ap[d] = f2bf(v);
}

// B (32x16 bf16 tile): lane = (n&15) | (((kk>>3)&1)<<4), elem j = (kk&7) | ((kk>>4)<<3).
__global__ void k_pack_b(const float* __restrict__ W, unsigned short* __restrict__ Bp,
                         int KC, int nShift, int KT) {
  int i = blockIdx.x * blockDim.x + threadIdx.x;
  int Cout = 1 << nShift;
  int total = KC << nShift;
  if (i >= total) return;
  int kg = i >> nShift;
  int n = i & (Cout - 1);
  float v = W[((size_t)kg << nShift) | n];
  int kt = kg >> 5, kk = kg & 31;
  int lane = (n & 15) | (((kk >> 3) & 1) << 4);
  int j = (kk & 7) | ((kk >> 4) << 3);
  size_t d = (((size_t)(n >> 4) * KT + kt) * 32 + lane) * 16 + j;
  Bp[d] = f2bf(v);
}

// ------------------------------ WMMA GEMM ----------------------------------
// Block = 8 waves (wave32). Block tile 64(M) x 128(N); wave tile 16 x 64
// (4 accumulators of 16x16 f32). K-loop step 32 via v_wmma_f32_16x16x32_bf16.
// B tile (8KB per k-step, shared by all 8 waves) is streamed into LDS with
// async global->LDS copies, double-buffered against the WMMAs.
__global__ __launch_bounds__(256)
void k_wmma_gemm(const unsigned short* __restrict__ Ap,
                 const unsigned short* __restrict__ Bp,
                 const float* __restrict__ bias,
                 float* __restrict__ out, int nValid, int Cout, int KT) {
  __shared__ unsigned short sB[2][8 * 512];   // 2 x 8KB double buffer

  int tid = threadIdx.x;
  int wave = tid >> 5;
  int lane = tid & 31;
  int wm = wave & 3;              // 4 waves along M
  int wn = wave >> 2;             // 2 waves along N
  int mt    = blockIdx.x * 4 + wm;
  int ntBlk = blockIdx.y * 8;
  int nt0   = ntBlk + wn * 4;

  // cooperative async copy: thread t moves tile bytes [t*32, t*32+32).
  // Tile layout in LDS == fragment-major global layout: [ntLocal][512 bf16].
  int ntLocal  = tid >> 5;                 // (tid*32) >> 10
  int fragByte = (tid & 31) * 32;          // (tid*32) & 1023
  unsigned long long gBase =
      (unsigned long long)Bp + ((size_t)(ntBlk + ntLocal) * KT) * 1024 + fragByte;
  unsigned ldsOff0 = (unsigned)(size_t)((char*)&sB[0][0] + tid * 32);
  unsigned ldsOff1 = (unsigned)(size_t)((char*)&sB[1][0] + tid * 32);

  auto issueCopy = [&](int kt, int buf) {
    unsigned long long g = gBase + (unsigned long long)kt * 1024u;
    unsigned l = buf ? ldsOff1 : ldsOff0;
    asm volatile(
        "global_load_async_to_lds_b128 %0, %1, off\n\t"
        "global_load_async_to_lds_b128 %0, %1, off offset:16"
        :: "v"(l), "v"(g) : "memory");
  };

  issueCopy(0, 0);

  const unsigned short* aBase = Ap + (size_t)mt * KT * 512 + lane * 16;
  v8f acc0 = {}, acc1 = {}, acc2 = {}, acc3 = {};

  for (int kt = 0; kt < KT; ++kt) {
    int buf = kt & 1;
    asm volatile("s_wait_asynccnt 0" ::: "memory");
    __syncthreads();                       // buf ready; prior reads of buf^1 done
    if (kt + 1 < KT) issueCopy(kt + 1, buf ^ 1);

    v16bf a = *(const v16bf*)(aBase + (size_t)kt * 512);
    if (kt + 1 < KT) __builtin_prefetch(aBase + (size_t)(kt + 1) * 512, 0, 1);

    const unsigned short* bl = &sB[buf][(wn * 4) * 512 + lane * 16];
    v16bf b0 = *(const v16bf*)(bl + 0 * 512);
    v16bf b1 = *(const v16bf*)(bl + 1 * 512);
    v16bf b2 = *(const v16bf*)(bl + 2 * 512);
    v16bf b3 = *(const v16bf*)(bl + 3 * 512);

    acc0 = __builtin_amdgcn_wmma_f32_16x16x32_bf16(false, a, false, b0, (short)0, acc0, false, false);
    acc1 = __builtin_amdgcn_wmma_f32_16x16x32_bf16(false, a, false, b1, (short)0, acc1, false, false);
    acc2 = __builtin_amdgcn_wmma_f32_16x16x32_bf16(false, a, false, b2, (short)0, acc2, false, false);
    acc3 = __builtin_amdgcn_wmma_f32_16x16x32_bf16(false, a, false, b3, (short)0, acc3, false, false);
  }

  // C/D layout: VGPR r -> M = mt*16 + r + 8*(lane>=16); N = nt*16 + (lane&15)
  int mBase = mt * 16 + (lane >> 4) * 8;
  int nInTile = lane & 15;
  v8f accs[4] = {acc0, acc1, acc2, acc3};
#pragma unroll
  for (int j = 0; j < 4; ++j) {
    int n = (nt0 + j) * 16 + nInTile;
    float bv = bias[n];
#pragma unroll
    for (int r = 0; r < 8; ++r) {
      int m = mBase + r;
      if (m < nValid) {
        float v = accs[j][r] + bv;
        out[(size_t)m * Cout + n] = v > 0.f ? v : 0.f;
      }
    }
  }
}

// ------------------------------- pooling -----------------------------------
__global__ void k_pool(const float* __restrict__ h, const int* __restrict__ batch,
                       float* __restrict__ psum, float* __restrict__ pcnt, int N) {
  int i = blockIdx.x * blockDim.x + threadIdx.x;
  if (i >= N * 64) return;
  int n = i >> 6;
  int cg = i & 63;
  int g = batch[n];
  const float4 v = ((const float4*)(h + (size_t)n * HIDC))[cg];
  float* d = psum + (size_t)g * HIDC + cg * 4;
  atomicAdd(d + 0, v.x);
  atomicAdd(d + 1, v.y);
  atomicAdd(d + 2, v.z);
  atomicAdd(d + 3, v.w);
  if (cg == 0) atomicAdd(&pcnt[g], 1.f);
}

__global__ void k_head(const float* __restrict__ psum, const float* __restrict__ pcnt,
                       const float* __restrict__ Wout, const float* __restrict__ bout,
                       float* __restrict__ out) {
  int i = blockIdx.x * blockDim.x + threadIdx.x;
  if (i >= NG * OUTC) return;
  int g = i >> 7;
  int oc = i & (OUTC - 1);
  float inv = 1.f / fmaxf(pcnt[g], 1.f);
  float acc = bout[oc];
  for (int c = 0; c < HIDC; ++c)
    acc = fmaf(psum[g * HIDC + c] * inv, Wout[c * OUTC + oc], acc);
  out[i] = acc;
}

// ------------------------------- driver ------------------------------------
static inline unsigned nblk(long long n, int t) { return (unsigned)((n + t - 1) / t); }

extern "C" void kernel_launch(void* const* d_in, const int* in_sizes, int n_in,
                              void* d_out, int out_size, void* d_ws, size_t ws_size,
                              hipStream_t stream) {
  const float* x     = (const float*)d_in[0];
  const float* W0    = (const float*)d_in[1];
  const float* b0    = (const float*)d_in[2];
  const float* W1    = (const float*)d_in[3];
  const float* b1    = (const float*)d_in[4];
  const float* W2    = (const float*)d_in[5];
  const float* b2    = (const float*)d_in[6];
  const float* Wout  = (const float*)d_in[7];
  const float* bout  = (const float*)d_in[8];
  const int*   eidx  = (const int*)d_in[9];
  const int*   batch = (const int*)d_in[10];
  float* out = (float*)d_out;

  const int* src = eidx;
  const int* dst = eidx + EE;

  // workspace carve (256B aligned)
  char* ws = (char*)d_ws;
  size_t off = 0;
  auto carve = [&](size_t bytes) -> void* {
    void* p = ws + off;
    off = (off + bytes + 255) & ~(size_t)255;
    return p;
  };
  float* deg  = (float*)carve((size_t)NN * 4);
  float* wbuf = (float*)carve((size_t)EE * 4);
  float* t0   = (float*)carve((size_t)NPAD * HIDC * 4);
  float* t1   = (float*)carve((size_t)NPAD * HIDC * 4);
  float* ybuf = (float*)carve((size_t)NPAD * HIDC * 4);
  float* hA   = (float*)carve((size_t)NPAD * HIDC * 4);
  float* hB   = (float*)carve((size_t)NPAD * HIDC * 4);
  unsigned short* Ap = (unsigned short*)carve((size_t)MT_CNT * KT_MAX * 512 * 2);
  unsigned short* Bp = (unsigned short*)carve((size_t)(HIDC / 16) * KT_MAX * 512 * 2);
  float* psum = (float*)carve((size_t)NG * HIDC * 4);
  float* pcnt = (float*)carve((size_t)NG * 4);

  const int T = 256;

  // edge normalization (once)
  k_zero_f32<<<nblk(NN, T), T, 0, stream>>>(deg, NN);
  k_deg<<<nblk(EE, T), T, 0, stream>>>(src, deg, EE);
  k_edgew<<<nblk(EE, T), T, 0, stream>>>(src, dst, deg, wbuf, EE);

  auto run_layer = [&](const float* xin, int Cin, const float* Wl, const float* bl,
                       float* hout) {
    int cShift = (Cin == 128) ? 7 : 8;
    int c4Shift = cShift - 2;
    int KT = (KCHEB * Cin) / 32;
    int KC = KCHEB * Cin;
    long long packTot = (long long)NPAD << cShift;
    long long nodeTot = (long long)NN * Cin;
    long long propTot = (long long)EE << c4Shift;

    // pack weights (B) fragment-major
    k_pack_b<<<nblk((long long)KC * HIDC, T), T, 0, stream>>>(Wl, Bp, KC, 8, KT);

    // Tx_0 = x
    k_pack_a<<<nblk(packTot, T), T, 0, stream>>>(xin, Ap, NN, cShift, KT, 0);

    // Tx_1 = L_hat x
    k_zero_f32<<<nblk(nodeTot, T), T, 0, stream>>>(t1, (int)nodeTot);
    k_propagate<<<nblk(propTot, T), T, 0, stream>>>(xin, src, dst, wbuf, t1, EE, c4Shift);
    k_pack_a<<<nblk(packTot, T), T, 0, stream>>>(t1, Ap, NN, cShift, KT, Cin);

    const float* p2 = xin;   // Tx_{k-2}
    float* p1 = t1;          // Tx_{k-1}
    for (int k = 2; k < KCHEB; ++k) {
      k_zero_f32<<<nblk(nodeTot, T), T, 0, stream>>>(ybuf, (int)nodeTot);
      k_propagate<<<nblk(propTot, T), T, 0, stream>>>(p1, src, dst, wbuf, ybuf, EE, c4Shift);
      float* tn = (k == 2) ? t0 : (float*)p2;       // k>=3: p2 is a scratch buffer
      k_cheb<<<nblk(nodeTot, T), T, 0, stream>>>(ybuf, p2, tn, (int)nodeTot);
      k_pack_a<<<nblk(packTot, T), T, 0, stream>>>(tn, Ap, NN, cShift, KT, k * Cin);
      p2 = p1;
      p1 = tn;
    }

    // fused GEMM: [NPAD x K*Cin] @ [K*Cin x 256] + bias, ReLU
    dim3 grid(NPAD / 64, HIDC / 128);
    k_wmma_gemm<<<grid, 256, 0, stream>>>(Ap, Bp, bl, hout, NN, HIDC, KT);
  };

  run_layer(x,  INC,  W0, b0, hA);
  run_layer(hA, HIDC, W1, b1, hB);
  run_layer(hB, HIDC, W2, b2, hA);

  // global mean pool + head
  k_zero_f32<<<nblk(NG * HIDC, T), T, 0, stream>>>(psum, NG * HIDC);
  k_zero_f32<<<nblk(NG, T), T, 0, stream>>>(pcnt, NG);
  k_pool<<<nblk((long long)NN * 64, T), T, 0, stream>>>(hA, batch, psum, pcnt, NN);
  k_head<<<nblk(NG * OUTC, T), T, 0, stream>>>(psum, pcnt, Wout, bout, out);
}
